// GNet_10213432230367
// MI455X (gfx1250) — compile-verified, matrix operand order checked by reading, not run on hardware
//
#include <hip/hip_runtime.h>
#include <math.h>

#define NN 8192      // nodes
#define NE 262144    // edges
#define FD 128       // input feature dim
#define HD 32        // hidden dim

typedef __attribute__((ext_vector_type(16))) _Float16 v16h;
typedef __attribute__((ext_vector_type(8)))  float    v8f;
typedef __attribute__((ext_vector_type(4)))  float    v4f;

// ---------------- degree / norm ----------------
__global__ void k_deg_init(float* __restrict__ deg, float* __restrict__ vp) {
    int i = blockIdx.x * blockDim.x + threadIdx.x;
    if (i < NN) deg[i] = 1.0f;            // self loop
    if (i < HD) vp[i] = 0.0f;             // zero value-pool accumulator
}

__global__ void k_deg_scatter(const int* __restrict__ dst, float* __restrict__ deg) {
    int e = blockIdx.x * blockDim.x + threadIdx.x;
    if (e < NE) atomicAdd(&deg[dst[e]], 1.0f);
}

__global__ void k_deg_rsqrt(const float* __restrict__ deg, float* __restrict__ dinv) {
    int i = blockIdx.x * blockDim.x + threadIdx.x;
    if (i < NN) dinv[i] = rsqrtf(deg[i]);
}

// ---------------- WMMA GEMM: out[NN,32] = x[NN,K] @ W[32,K]^T ----------------
// MODE 0 (conv): out_xw = x@W^T, out_agg = out_xw * dinv[r]^2   (self-loop init fused)
// MODE 1 (lin):  out_xw = tanh(x@W^T + bias)
// 256 threads = 8 waves; each wave -> one 16-row tile, full 32 output columns.
template<int MODE>
__global__ __launch_bounds__(256)
void k_gemm_wmma(const float* __restrict__ x, const float* __restrict__ W,
                 const float* __restrict__ bias, const float* __restrict__ dinv,
                 float* __restrict__ out_xw, float* __restrict__ out_agg, int K) {
    const int lane  = threadIdx.x & 31;
    const int wave  = threadIdx.x >> 5;
    const int tileM = (blockIdx.x * 8 + wave) * 16;
    const int hi    = lane >> 4;     // half-wave: selects K sub-range per ISA layout
    const int mr    = lane & 15;

    v8f acc0 = {};                   // columns 0..15
    v8f acc1 = {};                   // columns 16..31

    const float* xrow = x + (size_t)(tileM + mr) * K;
    const float* w0   = W + (size_t)mr * K;          // B col n = mr
    const float* w1   = W + (size_t)(16 + mr) * K;   // B col n = 16+mr

    for (int kb = 0; kb < K; kb += 32) {
        const int k0 = kb + hi * 8;
        v16h a, b0, b1;
        #pragma unroll
        for (int j = 0; j < 8; ++j) {
            a[j]      = (_Float16)xrow[k0 + j];        // K = k0..k0+7
            a[8 + j]  = (_Float16)xrow[k0 + 16 + j];   // K = k0+16..k0+23
            b0[j]     = (_Float16)w0[k0 + j];
            b0[8 + j] = (_Float16)w0[k0 + 16 + j];
            b1[j]     = (_Float16)w1[k0 + j];
            b1[8 + j] = (_Float16)w1[k0 + 16 + j];
        }
        acc0 = __builtin_amdgcn_wmma_f32_16x16x32_f16(false, a, false, b0,
                                                      (short)0, acc0, false, false);
        acc1 = __builtin_amdgcn_wmma_f32_16x16x32_f16(false, a, false, b1,
                                                      (short)0, acc1, false, false);
    }

    // hoist per-lane bias (MODE 1 only) so the epilogue loop is branch-free
    float blo = 0.f, bhi = 0.f;
    if (MODE == 1) { blo = bias[mr]; bhi = bias[16 + mr]; }

    // C/D layout: VGPR v, lanes 0-15: M=v, lanes 16-31: M=v+8; N = lane&15 (+16 for acc1)
    #pragma unroll
    for (int v = 0; v < 8; ++v) {
        const int r = tileM + v + hi * 8;
        float o0 = acc0[v], o1 = acc1[v];
        if (MODE == 1) {
            o0 = tanhf(o0 + blo);
            o1 = tanhf(o1 + bhi);
            out_xw[(size_t)r * HD + mr]      = o0;
            out_xw[(size_t)r * HD + 16 + mr] = o1;
        } else {
            const float dr = dinv[r];
            const float n  = dr * dr;                  // self-loop norm dinv[i]^2
            out_xw[(size_t)r * HD + mr]       = o0;
            out_xw[(size_t)r * HD + 16 + mr]  = o1;
            out_agg[(size_t)r * HD + mr]      = o0 * n;
            out_agg[(size_t)r * HD + 16 + mr] = o1 * n;
        }
    }
}

// ---------------- aggregation ----------------
__global__ void k_edge_scatter(const int* __restrict__ src, const int* __restrict__ dst,
                               const float* __restrict__ dinv, const float* __restrict__ xw,
                               float* __restrict__ agg) {
    int t = blockIdx.x * blockDim.x + threadIdx.x;   // NE*HD threads (fits int32)
    int f = t & 31;
    int e = t >> 5;
    if (e < NE) {
        int s = src[e], d = dst[e];
        float v = xw[s * HD + f] * dinv[s] * dinv[d];
        atomicAdd(&agg[d * HD + f], v);
    }
}

__global__ void k_bias_tanh(const float* __restrict__ agg, const float* __restrict__ bias,
                            float* __restrict__ h) {
    int t = blockIdx.x * blockDim.x + threadIdx.x;
    if (t < NN * HD) h[t] = tanhf(agg[t] + bias[t & 31]);
}

// ---------------- policy head ----------------
__global__ void k_row_sum(const float* __restrict__ h, float* __restrict__ pooled) {
    int i = blockIdx.x * blockDim.x + threadIdx.x;
    if (i < NN) {
        const v4f* r = (const v4f*)(h + (size_t)i * HD);
        float s = 0.f;
        #pragma unroll
        for (int j = 0; j < HD / 4; ++j) {
            v4f v = r[j];
            s += v.x + v.y + v.z + v.w;
        }
        pooled[i] = s;
    }
}

// out[row] = mask ? dot(pooled, Wop[row,:]) + bop[row] : -1e32
// 256 threads = 8 waves; 1 row per wave; pooled cached in LDS.
// Wop (268 MB > 192 MB L2) is streamed with non-temporal b128 loads so it does
// not evict the L2-resident graph data; this kernel is the HBM-bandwidth floor.
__global__ __launch_bounds__(256)
void k_gemv_logits(const float* __restrict__ pooled, const float* __restrict__ Wop,
                   const float* __restrict__ bop, const unsigned char* __restrict__ mask,
                   float* __restrict__ out) {
    __shared__ v4f sp[NN / 4];       // 32 KB of the 320 KB LDS
    const v4f* pg = (const v4f*)pooled;
    for (int i = threadIdx.x; i < NN / 4; i += 256) sp[i] = pg[i];
    __syncthreads();

    const int wave = threadIdx.x >> 5, lane = threadIdx.x & 31;
    const int row  = blockIdx.x * 8 + wave;
    const v4f* wr  = (const v4f*)(Wop + (size_t)row * NN);

    float acc = 0.f;
    #pragma unroll 4
    for (int j = lane; j < NN / 4; j += 32) {
        v4f w = __builtin_nontemporal_load(&wr[j]);      // TH=NT b128 stream
        v4f p = sp[j];
        acc += w.x * p.x + w.y * p.y + w.z * p.z + w.w * p.w;
    }
    #pragma unroll
    for (int off = 16; off > 0; off >>= 1) acc += __shfl_down(acc, off, 32);
    if (lane == 0) {
        float r = acc + bop[row];
        out[row] = mask[row] ? r : -1e32f;
    }
}

// ---------------- value head ----------------
__global__ __launch_bounds__(256)
void k_col_sum(const float* __restrict__ h, float* __restrict__ vp) {
    __shared__ float red[8][HD];
    const int wave = threadIdx.x >> 5, f = threadIdx.x & 31;
    const int base = blockIdx.x * 128;         // 128 rows per block, 64 blocks
    float s = 0.f;
    #pragma unroll
    for (int k = 0; k < 16; ++k) s += h[(size_t)(base + wave + k * 8) * HD + f];
    red[wave][f] = s;
    __syncthreads();
    if (wave == 0) {
        float t = 0.f;
        #pragma unroll
        for (int w = 0; w < 8; ++w) t += red[w][f];
        atomicAdd(&vp[f], t);
    }
}

__global__ void k_value(const float* __restrict__ vp, const float* __restrict__ Wov,
                        const float* __restrict__ bov, float* __restrict__ out) {
    int lane = threadIdx.x;
    float a = vp[lane] * Wov[lane];
    #pragma unroll
    for (int off = 16; off > 0; off >>= 1) a += __shfl_down(a, off, 32);
    if (lane == 0) out[NN] = a + bov[0];
}

// ---------------- launcher ----------------
extern "C" void kernel_launch(void* const* d_in, const int* in_sizes, int n_in,
                              void* d_out, int out_size, void* d_ws, size_t ws_size,
                              hipStream_t stream) {
    const float* x_p = (const float*)d_in[0];
    const float* x_s = (const float*)d_in[1];
    const int*   ei  = (const int*)d_in[2];
    const unsigned char* mask = (const unsigned char*)d_in[3];
    const float* W1p = (const float*)d_in[4];  const float* b1p = (const float*)d_in[5];
    const float* W2p = (const float*)d_in[6];  const float* b2p = (const float*)d_in[7];
    const float* Wlp = (const float*)d_in[8];  const float* blp = (const float*)d_in[9];
    const float* Wop = (const float*)d_in[10]; const float* bop = (const float*)d_in[11];
    const float* W1v = (const float*)d_in[12]; const float* b1v = (const float*)d_in[13];
    const float* W2v = (const float*)d_in[14]; const float* b2v = (const float*)d_in[15];
    const float* Wlv = (const float*)d_in[16]; const float* blv = (const float*)d_in[17];
    const float* Wov = (const float*)d_in[18]; const float* bov = (const float*)d_in[19];
    float* out = (float*)d_out;

    float* ws     = (float*)d_ws;
    float* deg    = ws;                             // NN
    float* dinv   = ws + (size_t)NN;                // NN
    float* pooled = ws + (size_t)2 * NN;            // NN
    float* vp     = ws + (size_t)3 * NN;            // 32 (padded to NN)
    float* b0     = ws + (size_t)4 * NN;            // NN*32
    float* b1     = ws + (size_t)36 * NN;           // NN*32
    float* b2     = ws + (size_t)68 * NN;           // NN*32

    const int* src = ei;
    const int* dst = ei + NE;

    // normalization terms
    k_deg_init   <<<NN / 256, 256, 0, stream>>>(deg, vp);
    k_deg_scatter<<<NE / 256, 256, 0, stream>>>(dst, deg);
    k_deg_rsqrt  <<<NN / 256, 256, 0, stream>>>(deg, dinv);

    auto run_trunk = [&](const float* x,
                         const float* W1, const float* bb1,
                         const float* W2, const float* bb2,
                         const float* Wl, const float* bl) {
        // conv1: xw (+fused self-loop agg init) -> edge aggregate -> +b, tanh
        k_gemm_wmma<0><<<NN / 128, 256, 0, stream>>>(x, W1, nullptr, dinv, b0, b1, FD);
        k_edge_scatter<<<(NE * HD) / 256, 256, 0, stream>>>(src, dst, dinv, b0, b1);
        k_bias_tanh   <<<(NN * HD) / 256, 256, 0, stream>>>(b1, bb1, b2);
        // conv2
        k_gemm_wmma<0><<<NN / 128, 256, 0, stream>>>(b2, W2, nullptr, dinv, b0, b1, HD);
        k_edge_scatter<<<(NE * HD) / 256, 256, 0, stream>>>(src, dst, dinv, b0, b1);
        k_bias_tanh   <<<(NN * HD) / 256, 256, 0, stream>>>(b1, bb2, b2);
        // linear + bias + tanh (fused in WMMA epilogue) -> b0
        k_gemm_wmma<1><<<NN / 128, 256, 0, stream>>>(b2, Wl, bl, nullptr, b0, nullptr, HD);
    };

    // ---- policy ----
    run_trunk(x_p, W1p, b1p, W2p, b2p, Wlp, blp);
    k_row_sum    <<<NN / 256, 256, 0, stream>>>(b0, pooled);
    k_gemv_logits<<<NN / 8, 256, 0, stream>>>(pooled, Wop, bop, mask, out);

    // ---- value ----
    run_trunk(x_s, W1v, b1v, W2v, b2v, Wlv, blv);
    k_col_sum<<<64, 256, 0, stream>>>(b0, vp);
    k_value  <<<1, 32, 0, stream>>>(vp, Wov, bov, out);
}
